// SelectivePMAC_2465311228010
// MI455X (gfx1250) — matrix operand order repaired and checked by reading, not compiled, not run on hardware
//
#include <hip/hip_runtime.h>
#include <math.h>

// ---------------------------------------------------------------------------
// SelectivePMAC for MI455X (gfx1250, wave32, WMMA).
// B=256, A=16, D=768, H=512.  ROWS = B*A = 4096.
//
// Algebraic fusion: influence = (sum_j gate*hc) @ W_c2 + gatesum*b_c2
// -> big GEMM shrinks from (65536x512)@(512x768) to (4096x512)@(512x768).
// GEMMs run on v_wmma_f32_16x16x32_bf16 with hi/lo bf16 splitting of f32
// operands (3 WMMAs per 16x16x32 chunk: ah*bh + ah*bl + al*bh) for
// near-f32 accuracy at bf16 matrix-core rate.  The split is precomputed
// once (prep kernel / k_gate) so GEMM inner loops are loads + WMMA only.
// ---------------------------------------------------------------------------

typedef __attribute__((ext_vector_type(16))) __bf16 v16bf;
typedef __attribute__((ext_vector_type(8)))  __bf16 v8bf;
typedef __attribute__((ext_vector_type(8)))  float  v8f;

#define NB 256
#define NA 16
#define ND 768
#define NH 512
#define NROWS (NB * NA)          // 4096

// --- bf16 fragment loaders (ISA 7.12.2 16-bit layouts, wave32) --------------

// A (16x32 K-chunk) from row-major bf16, leading dim ld (elements).
// lane l: M = l&15 ; element 2v+h: K = (v<4?0:16) + (l<16?0:8) + (v&3)*2 + h
// -> per lane: two aligned 16-byte runs of 8 consecutive K values.
static __device__ inline v16bf load_a_bf(const __bf16* __restrict__ base, int ld, int k0) {
  const int l  = threadIdx.x & 31;
  const int kk = (l < 16) ? 0 : 8;
  const __bf16* p = base + (l & 15) * ld + k0 + kk;
  const v8bf r0 = *(const v8bf*)p;          // K = k0+kk .. +7
  const v8bf r1 = *(const v8bf*)(p + 16);   // K = k0+16+kk .. +7
  return __builtin_shufflevector(r0, r1, 0, 1, 2, 3, 4, 5, 6, 7,
                                         8, 9, 10, 11, 12, 13, 14, 15);
}

// B (32x16 K-chunk) from row-major bf16 B[k][n]: lane l holds N=n0+(l&15),
// element s: K = k0 + (s<8 ? 0 : 16) + (l<16 ? 0 : 8) + (s&7) (strided gather).
static __device__ inline v16bf load_b_bf(const __bf16* __restrict__ base, int ld,
                                         int k0, int n0) {
  const int l = threadIdx.x & 31;
  const int kk = (l < 16) ? 0 : 8;
  const __bf16* p = base + (k0 + kk) * ld + n0 + (l & 15);
  v16bf r;
#pragma unroll
  for (int s = 0; s < 8; ++s) {
    r[s]     = p[s * ld];
    r[s + 8] = p[(16 + s) * ld];
  }
  return r;
}

static __device__ inline v8f wmma_split(const v16bf& ah, const v16bf& al,
                                        const v16bf& bh, const v16bf& bl, v8f c) {
  c = __builtin_amdgcn_wmma_f32_16x16x32_bf16(false, ah, false, bh, (short)0, c, false, false);
  c = __builtin_amdgcn_wmma_f32_16x16x32_bf16(false, ah, false, bl, (short)0, c, false, false);
  c = __builtin_amdgcn_wmma_f32_16x16x32_bf16(false, al, false, bh, (short)0, c, false, false);
  return c;
}

// ---------------------------------------------------------------------------
// Prep: split f32 array into bf16 hi/lo (x = hi + lo to ~bf16^2 precision).
// ---------------------------------------------------------------------------
__global__ __launch_bounds__(256) void k_split(const float* __restrict__ src,
                                               __bf16* __restrict__ hi,
                                               __bf16* __restrict__ lo, int n) {
  const int idx = blockIdx.x * 256 + threadIdx.x;
  if (idx < n) {
    const float x  = src[idx];
    const __bf16 h = (__bf16)x;
    hi[idx] = h;
    lo[idx] = (__bf16)(x - (float)h);
  }
}

// ---------------------------------------------------------------------------
// Kernel 1: proj = aspects(4096x768) @ [Wg_i|Wg_j|Wc_i|Wc_j](768x2048),
// stored as 4 contiguous (4096x512) f32 blocks: gi, gj, ci, cj.
// Wave tile = 16x64 (4 accumulators share one A fragment).
// 256 row-tiles x 32 col-groups = 8192 wave tiles; 8 waves/block -> 1024 blk.
// ---------------------------------------------------------------------------
__global__ __launch_bounds__(256) void k_gemm1(const __bf16* __restrict__ Ahi,
                                               const __bf16* __restrict__ Alo,
                                               const __bf16* __restrict__ Wg1hi,
                                               const __bf16* __restrict__ Wg1lo,
                                               const __bf16* __restrict__ Wc1hi,
                                               const __bf16* __restrict__ Wc1lo,
                                               float* __restrict__ proj) {
  const int id      = blockIdx.x * 8 + (threadIdx.x >> 5);
  const int rowTile = id & 255;                 // 0..255
  const int cg      = id >> 8;                  // 0..31 col-group of 64
  const int blk     = cg >> 3;                  // 0..3 : gi,gj,ci,cj
  const int h0      = (cg & 7) * 64;            // col offset within H
  const int woff    = (blk & 1) ? (ND * NH) : 0;
  const __bf16* Whi = (blk < 2 ? Wg1hi : Wc1hi) + woff;
  const __bf16* Wlo = (blk < 2 ? Wg1lo : Wc1lo) + woff;
  const __bf16* Ah  = Ahi + (rowTile * 16) * ND;
  const __bf16* Al  = Alo + (rowTile * 16) * ND;

  v8f acc[4] = {};
  for (int k0 = 0; k0 < ND; k0 += 32) {
    __builtin_prefetch(Whi + (k0 + 32) * NH + h0, 0, 1);
    const v16bf ah = load_a_bf(Ah, ND, k0);
    const v16bf al = load_a_bf(Al, ND, k0);
#pragma unroll
    for (int t = 0; t < 4; ++t) {
      const v16bf bh = load_b_bf(Whi, NH, k0, h0 + 16 * t);
      const v16bf bl = load_b_bf(Wlo, NH, k0, h0 + 16 * t);
      acc[t] = wmma_split(ah, al, bh, bl, acc[t]);
    }
  }

  float* out  = proj + blk * (NROWS * NH);
  const int l = threadIdx.x & 31;
#pragma unroll
  for (int t = 0; t < 4; ++t) {
    const int col = h0 + 16 * t + (l & 15);
#pragma unroll
    for (int j = 0; j < 8; ++j) {
      const int row = rowTile * 16 + ((l < 16) ? j : 8 + j);
      out[row * NH + col] = acc[t][j];
    }
  }
}

// ---------------------------------------------------------------------------
// Kernel 2: per batch b: gate (16x16, sigmoid, zero diag), gatesum, and
// M[b,i,h] = sum_j gate[b,i,j]*relu(ci[b,i,h]+cj[b,j,h]+b_c1[h]),
// written directly as bf16 hi/lo for the next GEMM.
// ---------------------------------------------------------------------------
__global__ __launch_bounds__(256) void k_gate(const float* __restrict__ gi,
                                              const float* __restrict__ gj,
                                              const float* __restrict__ ci,
                                              const float* __restrict__ cj,
                                              const float* __restrict__ b_g1,
                                              const float* __restrict__ w_g2,
                                              const float* __restrict__ b_g2,
                                              const float* __restrict__ b_c1,
                                              float* __restrict__ gate_out,
                                              __bf16* __restrict__ Mhi,
                                              __bf16* __restrict__ Mlo,
                                              float* __restrict__ gatesum) {
  const int b = blockIdx.x;
  __shared__ float sg[16][16];
  const int i = threadIdx.x >> 4;
  const int j = threadIdx.x & 15;

  // gate[i][j]
  const float* gri = gi + (b * 16 + i) * NH;
  const float* grj = gj + (b * 16 + j) * NH;
  float acc = 0.f;
  for (int h = 0; h < NH; ++h)
    acc += fmaxf(gri[h] + grj[h] + b_g1[h], 0.f) * w_g2[h];
  float g = 0.f;
  if (i != j) g = 1.f / (1.f + __expf(-(acc + b_g2[0])));
  sg[i][j] = g;
  gate_out[b * 256 + i * 16 + j] = g;
  __syncthreads();

  if (threadIdx.x < 16) {
    float s = 0.f;
#pragma unroll
    for (int jj = 0; jj < 16; ++jj) s += sg[threadIdx.x][jj];
    gatesum[b * 16 + threadIdx.x] = s;
  }
  __syncthreads();

  // M[i][h]: thread (i, j) covers h = j, j+16, ...
  const float* cri = ci + (b * 16 + i) * NH;
  for (int h = j; h < NH; h += 16) {
    const float base = cri[h] + b_c1[h];
    float m = 0.f;
#pragma unroll
    for (int jj = 0; jj < 16; ++jj)
      m += sg[i][jj] * fmaxf(base + cj[(b * 16 + jj) * NH + h], 0.f);
    const int idx  = (b * 16 + i) * NH + h;
    const __bf16 mh = (__bf16)m;
    Mhi[idx] = mh;
    Mlo[idx] = (__bf16)(m - (float)mh);
  }
}

// ---------------------------------------------------------------------------
// Kernel 3: influence = M(4096x512) @ W_c2(512x768), fused epilogue:
// final = aspects + influence + gatesum*b_c2, then LayerNorm + mask.
// One block per row-tile (== batch); wave w owns cols [w*96, w*96+96):
// 6 accumulators sharing one A fragment per K-chunk.  16x768 f32 row-tile
// staged in LDS for the LN.
// ---------------------------------------------------------------------------
__global__ __launch_bounds__(256) void k_gemm2_ln(const __bf16* __restrict__ Mhi,
                                                  const __bf16* __restrict__ Mlo,
                                                  const __bf16* __restrict__ Whi,
                                                  const __bf16* __restrict__ Wlo,
                                                  const float* __restrict__ b_c2,
                                                  const float* __restrict__ gatesum,
                                                  const float* __restrict__ X,
                                                  const float* __restrict__ gamma,
                                                  const float* __restrict__ beta,
                                                  const unsigned char* __restrict__ mask,
                                                  float* __restrict__ out) {
  const int rt = blockIdx.x;                 // row tile == batch index
  __shared__ float tile[16][ND];
  __shared__ float red1[16][16];
  __shared__ float red2[16][16];
  __shared__ float smu[16], srs[16];

  const int w  = threadIdx.x >> 5;
  const int l  = threadIdx.x & 31;
  const int n0 = w * 96;
  const __bf16* Ah = Mhi + rt * 16 * NH;
  const __bf16* Al = Mlo + rt * 16 * NH;

  v8f acc[6] = {};
  for (int k0 = 0; k0 < NH; k0 += 32) {
    __builtin_prefetch(Whi + (k0 + 32) * ND + n0, 0, 1);
    const v16bf ah = load_a_bf(Ah, NH, k0);
    const v16bf al = load_a_bf(Al, NH, k0);
#pragma unroll
    for (int t = 0; t < 6; ++t) {
      const v16bf bh = load_b_bf(Whi, ND, k0, n0 + 16 * t);
      const v16bf bl = load_b_bf(Wlo, ND, k0, n0 + 16 * t);
      acc[t] = wmma_split(ah, al, bh, bl, acc[t]);
    }
  }

#pragma unroll
  for (int t = 0; t < 6; ++t) {
    const int col = n0 + 16 * t + (l & 15);
#pragma unroll
    for (int jj = 0; jj < 8; ++jj) {
      const int r    = (l < 16) ? jj : 8 + jj;
      const int grow = rt * 16 + r;
      tile[r][col] = acc[t][jj] + gatesum[grow] * b_c2[col] + X[grow * ND + col];
    }
  }
  __syncthreads();

  // LayerNorm over D=768 per row: 16 threads per row, 48 elems each.
  const int r  = threadIdx.x >> 4;
  const int c0 = threadIdx.x & 15;
  float s = 0.f, s2 = 0.f;
  for (int c = c0; c < ND; c += 16) { const float x = tile[r][c]; s += x; s2 += x * x; }
  red1[r][c0] = s; red2[r][c0] = s2;
  __syncthreads();
  if (c0 == 0) {
    float ts = 0.f, ts2 = 0.f;
#pragma unroll
    for (int k = 0; k < 16; ++k) { ts += red1[r][k]; ts2 += red2[r][k]; }
    const float mu  = ts / (float)ND;
    const float var = ts2 / (float)ND - mu * mu;
    smu[r] = mu;
    srs[r] = rsqrtf(var + 1e-5f);
  }
  __syncthreads();
  const float mu = smu[r], rs = srs[r];
  const float mk = mask[rt * 16 + r] ? 1.f : 0.f;
  for (int c = c0; c < ND; c += 16) {
    const float y = ((tile[r][c] - mu) * rs * gamma[c] + beta[c]) * mk;
    out[(rt * 16 + r) * ND + c] = y;
  }
}

// ---------------------------------------------------------------------------
extern "C" void kernel_launch(void* const* d_in, const int* in_sizes, int n_in,
                              void* d_out, int out_size, void* d_ws, size_t ws_size,
                              hipStream_t stream) {
  const float* aspects      = (const float*)d_in[0];
  const unsigned char* mask = (const unsigned char*)d_in[1];
  const float* W_g1         = (const float*)d_in[2];
  const float* b_g1         = (const float*)d_in[3];
  const float* w_g2         = (const float*)d_in[4];
  const float* b_g2         = (const float*)d_in[5];
  const float* W_c1         = (const float*)d_in[6];
  const float* b_c1         = (const float*)d_in[7];
  const float* W_c2         = (const float*)d_in[8];
  const float* b_c2         = (const float*)d_in[9];
  const float* ln_gamma     = (const float*)d_in[10];
  const float* ln_beta      = (const float*)d_in[11];

  float* out_final = (float*)d_out;                 // (B*A, D)
  float* out_gate  = out_final + NROWS * ND;        // (B, A, A)

  // workspace layout
  float* ws   = (float*)d_ws;
  float* proj = ws;                                 // 4 * 4096*512 f32
  float* gsum = proj + 4 * (NROWS * NH);            // 4096 f32
  __bf16* b16 = (__bf16*)(gsum + NROWS);
  __bf16* Ahi   = b16;                 b16 += NROWS * ND;
  __bf16* Alo   = b16;                 b16 += NROWS * ND;
  __bf16* Wg1hi = b16;                 b16 += 2 * ND * NH;
  __bf16* Wg1lo = b16;                 b16 += 2 * ND * NH;
  __bf16* Wc1hi = b16;                 b16 += 2 * ND * NH;
  __bf16* Wc1lo = b16;                 b16 += 2 * ND * NH;
  __bf16* Wc2hi = b16;                 b16 += NH * ND;
  __bf16* Wc2lo = b16;                 b16 += NH * ND;
  __bf16* Mhi   = b16;                 b16 += NROWS * NH;
  __bf16* Mlo   = b16;

  // Split f32 -> bf16 hi/lo once.
  k_split<<<(NROWS * ND + 255) / 256, 256, 0, stream>>>(aspects, Ahi, Alo, NROWS * ND);
  k_split<<<(2 * ND * NH + 255) / 256, 256, 0, stream>>>(W_g1, Wg1hi, Wg1lo, 2 * ND * NH);
  k_split<<<(2 * ND * NH + 255) / 256, 256, 0, stream>>>(W_c1, Wc1hi, Wc1lo, 2 * ND * NH);
  k_split<<<(NH * ND + 255) / 256, 256, 0, stream>>>(W_c2, Wc2hi, Wc2lo, NH * ND);

  k_gemm1<<<1024, 256, 0, stream>>>(Ahi, Alo, Wg1hi, Wg1lo, Wc1hi, Wc1lo, proj);
  k_gate<<<NB, 256, 0, stream>>>(proj,
                                 proj + 1 * (NROWS * NH),
                                 proj + 2 * (NROWS * NH),
                                 proj + 3 * (NROWS * NH),
                                 b_g1, w_g2, b_g2, b_c1,
                                 out_gate, Mhi, Mlo, gsum);
  k_gemm2_ln<<<NB, 256, 0, stream>>>(Mhi, Mlo, Wc2hi, Wc2lo, b_c2, gsum, aspects,
                                     ln_gamma, ln_beta, mask, out_final);
}
